// Attention_23811298689437
// MI455X (gfx1250) — matrix operand order repaired and checked by reading, not compiled, not run on hardware
//
#include <hip/hip_runtime.h>
#include <hip/hip_bf16.h>

// ----------------------------------------------------------------------------
// ViT windowed attention with decomposed rel-pos, MI455X (gfx1250, wave32).
// Pipeline: cvt/transpose -> WMMA QKV GEMM -> rel bias -> fused flash attn
//           (WMMA QK^T + online softmax + WMMA PV) -> WMMA out-proj.
// Matmuls: v_wmma_f32_16x16x32_f16.  GEMM staging: double-buffered
// global_load_async_to_lds_b128 (ASYNCcnt).  Attention K/V staging: TDM
// tensor_load_to_lds (TENSORcnt) with hardware LDS padding, when the
// builtin is available; async-to-LDS fallback otherwise.
// ----------------------------------------------------------------------------

typedef __attribute__((ext_vector_type(16))) _Float16 v16h;
typedef __attribute__((ext_vector_type(8)))  _Float16 v8h;
typedef __attribute__((ext_vector_type(8)))  float    v8f;

#define NH  12
#define HD  64
#define Bb  8
#define Ss  1024
#define Cc  768
#define BHh 96

static __device__ __forceinline__ v16h pack16(v8h a, v8h b) {
  union { v16h v; v8h h[2]; } u;
  u.h[0] = a; u.h[1] = b;
  return u.v;
}
static __device__ __forceinline__ v8f wmma_f16(v16h a, v16h b, v8f c) {
  // (neg_a, A, neg_b, B, c_mod, C, reuse_a, reuse_b)
  return __builtin_amdgcn_wmma_f32_16x16x32_f16(false, a, false, b, (short)0, c,
                                                false, false);
}
static __device__ __forceinline__ v8f zero8() {
  v8f z = {0.f, 0.f, 0.f, 0.f, 0.f, 0.f, 0.f, 0.f};
  return z;
}

// ---- async global->LDS staging (ASYNCcnt).  offset: applies to BOTH the
// LDS destination and the global source address (ISA 08_async_tensor §4.4).
static __device__ __forceinline__ void async_ld16(const void* g, void* l) {
  unsigned lo = (unsigned)(size_t)l;  // low 32 bits of generic = LDS offset
  asm volatile("global_load_async_to_lds_b128 %0, %1, off"
               :: "v"(lo), "v"(g) : "memory");
}
static __device__ __forceinline__ void async_ld32(const void* g, void* l) {
  unsigned lo = (unsigned)(size_t)l;
  asm volatile("global_load_async_to_lds_b128 %0, %1, off\n\t"
               "global_load_async_to_lds_b128 %0, %1, off offset:16"
               :: "v"(lo), "v"(g) : "memory");
}
template <int N>
static __device__ __forceinline__ void wait_async() {
#if __has_builtin(__builtin_amdgcn_s_wait_asynccnt)
  __builtin_amdgcn_s_wait_asynccnt(N);
#else
  if constexpr (N == 0) asm volatile("s_wait_asynccnt 0x0" ::: "memory");
  else if constexpr (N == 3) asm volatile("s_wait_asynccnt 0x3" ::: "memory");
  else asm volatile("s_wait_asynccnt 0x4" ::: "memory");
#endif
}

// ---- Tensor Data Mover: 2D tile global->LDS with LDS padding --------------
#if __has_builtin(__builtin_amdgcn_tensor_load_to_lds) && \
    __has_builtin(__builtin_amdgcn_s_wait_tensorcnt)
#define HAVE_TDM 1
typedef __attribute__((ext_vector_type(4))) unsigned tdm_u32x4;
typedef __attribute__((ext_vector_type(8))) int      tdm_i32x8;
typedef __attribute__((ext_vector_type(4))) int      tdm_i32x4;

// tile_x contiguous elements per row, tile_y rows, stride = row pitch in
// elements (data_size = 2B).  pad_i: pad every 2^(pad_i+1) DWORDs;
// pad_a: insert (pad_a+1) DWORDs of LDS padding (ISA 08 §8.3/8.4).
static __device__ __forceinline__ void tdm_load_2d(
    unsigned lds_addr, const void* gaddr, unsigned tile_x, unsigned tile_y,
    unsigned stride, unsigned pad_i, unsigned pad_a) {
  unsigned long long ga = (unsigned long long)(size_t)gaddr;
  tdm_u32x4 g0;
  g0.x = 1u;                                           // count=1 (valid D#)
  g0.y = lds_addr;                                     // lds_addr[31:0]
  g0.z = (unsigned)ga;                                 // global_addr[31:0]
  g0.w = ((unsigned)(ga >> 32) & 0x01FFFFFFu) | 0x80000000u;  // addr|type=2
  tdm_i32x8 g1;
  g1[0] = (1 << 16) | (1 << 20) | (int)(pad_i << 22) | (int)(pad_a << 25);
  g1[1] = (int)((tile_x & 0xFFFFu) << 16);             // tensor_dim0 = tile_x
  g1[2] = (int)((tile_y & 0xFFFFu) << 16);             // tensor_dim1 = tile_y
  g1[3] = (int)((tile_x & 0xFFFFu) << 16);             // tile_dim0
  g1[4] = (int)(tile_y & 0xFFFFu);                     // tile_dim1 (tile_dim2=0)
  g1[5] = (int)stride;                                 // tensor_dim0_stride
  g1[6] = 0; g1[7] = 0;
  tdm_i32x4 z4 = {0, 0, 0, 0};
#if __has_include(<hip/amd_detail/amd_gfx1250_TDM.h>)
  tdm_i32x8 z8 = {0, 0, 0, 0, 0, 0, 0, 0};             // amdgpu-toolchain: 6 args
  __builtin_amdgcn_tensor_load_to_lds(g0, g1, z4, z4, z8, 0);
#else
  __builtin_amdgcn_tensor_load_to_lds(g0, g1, z4, z4, 0);  // ROCm 7.2: 5 args
#endif
}
#endif  // HAVE_TDM

// ---------------------------------------------------------------- conversions
__global__ void cvt_f16_kernel(const float* __restrict__ in,
                               _Float16* __restrict__ out, int n) {
  int i = blockIdx.x * blockDim.x + threadIdx.x;
  if (i < n) out[i] = (_Float16)in[i];
}

// in[rows][cols] (f32) -> out[cols][rows] (f16)
__global__ void transpose_cvt_kernel(const float* __restrict__ in,
                                     _Float16* __restrict__ out,
                                     int rows, int cols) {
  int i = blockIdx.x * blockDim.x + threadIdx.x;
  if (i >= rows * cols) return;
  int r = i / cols, c = i - r * cols;
  out[(size_t)c * rows + r] = (_Float16)in[i];
}

// ------------------------------------------------------------- WMMA GEMM
// C[M][N] = A[M][K] * BT[N][K]^T + bias.  Block tile 128x64, 8 waves,
// each wave owns a 64x16 tile = 4 C fragments.  Double-buffered async
// LDS staging.  mode 0: f32 row-major out.  mode 1: scatter q/k/v^T.
__global__ __launch_bounds__(256) void wmma_gemm_kernel(
    const _Float16* __restrict__ A, const _Float16* __restrict__ BT,
    const float* __restrict__ bias, int M, int N, int K, int mode,
    float* __restrict__ outF, _Float16* __restrict__ qh,
    _Float16* __restrict__ kh, _Float16* __restrict__ vT) {
  __shared__ _Float16 As[2][128][40];
  __shared__ _Float16 Bs[2][64][40];

  const int tid  = threadIdx.x;
  const int lane = tid & 31, wid = tid >> 5;
  const int half = lane >> 4, r = lane & 15;
  const int m0 = blockIdx.x * 128, n0 = blockIdx.y * 64;
  const int mBase = (wid >> 2) * 64, nBase = (wid & 3) * 16;

  v8f acc[4] = {zero8(), zero8(), zero8(), zero8()};

  const int aRow = tid >> 1, aCol = (tid & 1) * 16;
  const int bRow = tid >> 2, bCol = (tid & 3) * 8;
  const _Float16* aG = A + (size_t)(m0 + aRow) * K + aCol;
  const _Float16* bG = BT + (size_t)(n0 + bRow) * K + bCol;
  const int KT = K / 32;

  // stage tile 0
  async_ld32(aG, &As[0][aRow][aCol]);
  async_ld16(bG, &Bs[0][bRow][bCol]);

  int cur = 0;
  for (int kt = 0; kt < KT; ++kt) {
    if (kt + 2 < KT)  // warm L2 two tiles ahead
      __builtin_prefetch(aG + (kt + 2) * 32, 0, 0);
    if (kt + 1 < KT) {
      async_ld32(aG + (kt + 1) * 32, &As[cur ^ 1][aRow][aCol]);
      async_ld16(bG + (kt + 1) * 32, &Bs[cur ^ 1][bRow][bCol]);
      wait_async<3>();  // in-order completion: tile kt is now in LDS
    } else {
      wait_async<0>();
    }
    __syncthreads();

    v16h bf = pack16(*(const v8h*)&Bs[cur][nBase + r][half * 8],
                     *(const v8h*)&Bs[cur][nBase + r][16 + half * 8]);
    v16h af[4];
#pragma unroll
    for (int ms = 0; ms < 4; ++ms)
      af[ms] = pack16(*(const v8h*)&As[cur][mBase + ms * 16 + r][half * 8],
                      *(const v8h*)&As[cur][mBase + ms * 16 + r][16 + half * 8]);
#pragma unroll
    for (int ms = 0; ms < 4; ++ms) acc[ms] = wmma_f16(af[ms], bf, acc[ms]);

    __syncthreads();  // buffer we just read gets refilled next iteration
    cur ^= 1;
  }

  const int nGlob = n0 + nBase + r;
  const float bn = bias[nGlob];
#pragma unroll
  for (int ms = 0; ms < 4; ++ms) {
#pragma unroll
    for (int e = 0; e < 8; ++e) {
      int mGlob = m0 + mBase + ms * 16 + half * 8 + e;
      float val = acc[ms][e] + bn;
      if (mode == 0) {
        outF[(size_t)mGlob * N + nGlob] = val;
      } else {
        int which = nGlob / Cc;
        int cc = nGlob - which * Cc;
        int head = cc >> 6, d = cc & 63;
        int b = mGlob >> 10, s = mGlob & 1023;
        int bh = b * NH + head;
        if (which == 0)
          qh[((size_t)bh * Ss + s) * HD + d] = (_Float16)(val * 0.125f);
        else if (which == 1)
          kh[((size_t)bh * Ss + s) * HD + d] = (_Float16)val;
        else
          vT[((size_t)bh * HD + d) * Ss + s] = (_Float16)val;
      }
    }
  }
}

// ------------------------------------------------------- rel-pos bias dots
// relh[bh][q][j] = (q_vec . rel_pos_h[qh - j + 31]),  relw likewise with qw.
// qh is stored pre-scaled by 1/8 -> multiply result back by 8.
__global__ void rel_bias_kernel(const _Float16* __restrict__ qh,
                                const float* __restrict__ rph,
                                const float* __restrict__ rpw,
                                float* __restrict__ relh,
                                float* __restrict__ relw) {
  int idx = blockIdx.x * blockDim.x + threadIdx.x;  // [0, 96*1024*32)
  int bh = idx >> 15;
  int rem = idx & 32767;
  int qpos = rem >> 5;
  int j = rem & 31;
  int qrow = qpos >> 5, qcol = qpos & 31;
  const _Float16* qp = qh + ((size_t)bh * Ss + qpos) * HD;
  const float* ph = rph + (size_t)(qrow - j + 31) * HD;
  const float* pw = rpw + (size_t)(qcol - j + 31) * HD;
  float ah = 0.f, aw = 0.f;
#pragma unroll 8
  for (int d = 0; d < HD; ++d) {
    float qv = (float)qp[d];
    ah += qv * ph[d];
    aw += qv * pw[d];
  }
  relh[((size_t)bh * Ss + qpos) * 32 + j] = ah * 8.0f;
  relw[((size_t)bh * Ss + qpos) * 32 + j] = aw * 8.0f;
}

// ----------------------------------------------------- fused flash attention
// Grid (16 q-tiles, 96 bh). 128 threads = 4 waves; each wave owns 16 query
// rows, O accumulator 16x64 f32 (4 C frags). Keys streamed in 32-wide blocks;
// K/V tiles staged by the Tensor Data Mover (wave 0 issues, TENSORcnt) with
// hardware LDS padding matching the banked layouts, double-buffered.
__global__ __launch_bounds__(128) void flash_attn_kernel(
    const _Float16* __restrict__ qh, const _Float16* __restrict__ kh,
    const _Float16* __restrict__ vT, const float* __restrict__ relh,
    const float* __restrict__ relw, _Float16* __restrict__ oh) {
  __shared__ _Float16 Ks[2][32][72];    // 32 keys x 64 dims (+4 DW pad/row)
  __shared__ _Float16 Vs[2][64][40];    // 64 dims x 32 keys (+4 DW pad/row)
  __shared__ float    RelH[64][32];     // WG's 64 query rows x 32 key-rows
  __shared__ _Float16 Ps[4][16][40];    // per-wave P relayout (C -> A)

  const int tid = threadIdx.x;
  const int wave = tid >> 5, lane = tid & 31;
  const int half = lane >> 4, r = lane & 15;
  const int bh = blockIdx.y;
  const int qWG = blockIdx.x * 64;
  const int qwave = qWG + wave * 16;

  const _Float16* kBase = kh + (size_t)bh * Ss * HD;   // [key][64]
  const _Float16* vBase = vT + (size_t)bh * HD * Ss;   // [64][key]

#ifdef HAVE_TDM
  if (wave == 0) {  // TDM ignores EXEC; one wave issues DMA for the WG
    tdm_load_2d((unsigned)(size_t)&Ks[0][0][0], kBase, 64, 32, 64, 4, 3);
    tdm_load_2d((unsigned)(size_t)&Vs[0][0][0], vBase, 32, 64, 1024, 3, 3);
  }
#else
  const int kRow = tid >> 2, kOff = (tid & 3) * 16;
  const int vRow = tid >> 1, vOff = (tid & 1) * 16;
  async_ld32(kBase + (size_t)kRow * HD + kOff, &Ks[0][kRow][kOff]);
  async_ld32(vBase + (size_t)vRow * Ss + vOff, &Vs[0][vRow][vOff]);
#endif

  // Q fragments (A layout), head dim split in two K=32 chunks
  v16h qf[2];
  {
    const _Float16* qrow = qh + ((size_t)bh * Ss + qwave + r) * HD;
#pragma unroll
    for (int kc = 0; kc < 2; ++kc) {
      const _Float16* p = qrow + kc * 32 + half * 8;
      qf[kc] = pack16(*(const v8h*)p, *(const v8h*)(p + 16));
    }
  }

  // rel_h slice for this WG (key-row index == blk since blocks are 32-aligned)
#pragma unroll
  for (int i = 0; i < 16; ++i) {
    int idx = i * 128 + tid;
    int ql = idx >> 5, kc = idx & 31;
    RelH[ql][kc] = relh[((size_t)bh * Ss + qWG + ql) * 32 + kc];
  }

  // rel_w: key-col index == ns*16 + r, constant across blocks -> registers
  float relw_reg[2][8];
#pragma unroll
  for (int ns = 0; ns < 2; ++ns) {
    int kw = ns * 16 + r;
#pragma unroll
    for (int e = 0; e < 8; ++e) {
      int ql = wave * 16 + half * 8 + e;
      relw_reg[ns][e] = relw[((size_t)bh * Ss + qWG + ql) * 32 + kw];
    }
  }

  float m_i[8], l_i[8];
  v8f of[4] = {zero8(), zero8(), zero8(), zero8()};
#pragma unroll
  for (int e = 0; e < 8; ++e) { m_i[e] = -1e30f; l_i[e] = 0.0f; }

  int cur = 0;
  for (int blk = 0; blk < 32; ++blk) {
    if (blk + 1 < 32) {
#ifdef HAVE_TDM
      if (wave == 0) {  // issue next block; wait for current (in-order)
        tdm_load_2d((unsigned)(size_t)&Ks[cur ^ 1][0][0],
                    kBase + (size_t)(blk + 1) * 32 * HD, 64, 32, 64, 4, 3);
        tdm_load_2d((unsigned)(size_t)&Vs[cur ^ 1][0][0],
                    vBase + (blk + 1) * 32, 32, 64, 1024, 3, 3);
        __builtin_amdgcn_s_wait_tensorcnt(2);
      }
#else
      async_ld32(kBase + ((size_t)(blk + 1) * 32 + kRow) * HD + kOff,
                 &Ks[cur ^ 1][kRow][kOff]);
      async_ld32(vBase + (size_t)vRow * Ss + (blk + 1) * 32 + vOff,
                 &Vs[cur ^ 1][vRow][vOff]);
      wait_async<4>();
#endif
    } else {
#ifdef HAVE_TDM
      if (wave == 0) __builtin_amdgcn_s_wait_tensorcnt(0);
#else
      wait_async<0>();
#endif
    }
    __syncthreads();

    // S = Q K^T for two 16-key subtiles
    v8f sf[2];
#pragma unroll
    for (int ns = 0; ns < 2; ++ns) {
      v8f a = zero8();
#pragma unroll
      for (int kc = 0; kc < 2; ++kc) {
        const _Float16* bp = &Ks[cur][ns * 16 + r][kc * 32 + half * 8];
        v16h bf = pack16(*(const v8h*)bp, *(const v8h*)(bp + 16));
        a = wmma_f16(qf[kc], bf, a);
      }
      sf[ns] = a;
    }

    // bias + online softmax (row = half*8+e, lives across 16 lanes)
    float corr[8];
#pragma unroll
    for (int e = 0; e < 8; ++e) {
      float rh = RelH[wave * 16 + half * 8 + e][blk];
      float s0 = sf[0][e] + rh + relw_reg[0][e];
      float s1 = sf[1][e] + rh + relw_reg[1][e];
      float mx = fmaxf(s0, s1);
      mx = fmaxf(mx, __shfl_xor(mx, 1));
      mx = fmaxf(mx, __shfl_xor(mx, 2));
      mx = fmaxf(mx, __shfl_xor(mx, 4));
      mx = fmaxf(mx, __shfl_xor(mx, 8));
      float mn = fmaxf(m_i[e], mx);
      float c = __expf(m_i[e] - mn);
      m_i[e] = mn;
      float p0 = __expf(s0 - mn);
      float p1 = __expf(s1 - mn);
      sf[0][e] = p0; sf[1][e] = p1;
      float rs = p0 + p1;
      rs += __shfl_xor(rs, 1);
      rs += __shfl_xor(rs, 2);
      rs += __shfl_xor(rs, 4);
      rs += __shfl_xor(rs, 8);
      l_i[e] = l_i[e] * c + rs;
      corr[e] = c;
    }
#pragma unroll
    for (int dn = 0; dn < 4; ++dn)
#pragma unroll
      for (int e = 0; e < 8; ++e) of[dn][e] *= corr[e];

    // P: C-layout -> LDS -> A-layout (wave-synchronous, DS in-order per wave)
#pragma unroll
    for (int ns = 0; ns < 2; ++ns)
#pragma unroll
      for (int e = 0; e < 8; ++e)
        Ps[wave][half * 8 + e][ns * 16 + r] = (_Float16)sf[ns][e];
    __builtin_amdgcn_wave_barrier();
    const _Float16* pr = &Ps[wave][r][0];
    v16h pf = pack16(*(const v8h*)(pr + half * 8),
                     *(const v8h*)(pr + 16 + half * 8));

    // O += P V
#pragma unroll
    for (int dn = 0; dn < 4; ++dn) {
      const _Float16* vb = &Vs[cur][dn * 16 + r][half * 8];
      v16h vf = pack16(*(const v8h*)vb, *(const v8h*)(vb + 16));
      of[dn] = wmma_f16(pf, vf, of[dn]);
    }

    __syncthreads();  // buffer we just read gets refilled next iteration
    cur ^= 1;
  }

  // normalize, store f16 in [b][s][head*64+d] for the out-proj GEMM
  const int b = bh / NH, head = bh - b * NH;
#pragma unroll
  for (int dn = 0; dn < 4; ++dn) {
    int d = head * 64 + dn * 16 + r;
#pragma unroll
    for (int e = 0; e < 8; ++e) {
      int q = qwave + half * 8 + e;
      oh[((size_t)b * Ss + q) * Cc + d] = (_Float16)(of[dn][e] / l_i[e]);
    }
  }
}

// ---------------------------------------------------------------- launcher
extern "C" void kernel_launch(void* const* d_in, const int* in_sizes, int n_in,
                              void* d_out, int out_size, void* d_ws,
                              size_t ws_size, hipStream_t stream) {
  const float* x     = (const float*)d_in[0];
  const float* wqkv  = (const float*)d_in[1];
  const float* bqkv  = (const float*)d_in[2];
  const float* rph   = (const float*)d_in[3];
  const float* rpw   = (const float*)d_in[4];
  const float* wproj = (const float*)d_in[5];
  const float* bproj = (const float*)d_in[6];
  float* out = (float*)d_out;

  char* ws = (char*)d_ws;
  size_t off = 0;
  auto alloc = [&](size_t bytes) {
    char* p = ws + off;
    off += (bytes + 255) & ~(size_t)255;
    return p;
  };
  const size_t M1 = (size_t)Bb * Ss;                 // 8192
  _Float16* xh  = (_Float16*)alloc(M1 * Cc * 2);     // x in f16
  _Float16* wqT = (_Float16*)alloc((size_t)3 * Cc * Cc * 2);  // [3C][C]
  _Float16* wpT = (_Float16*)alloc((size_t)Cc * Cc * 2);      // [C][C]
  _Float16* qf  = (_Float16*)alloc((size_t)BHh * Ss * HD * 2);
  _Float16* kf  = (_Float16*)alloc((size_t)BHh * Ss * HD * 2);
  _Float16* vTf = (_Float16*)alloc((size_t)BHh * HD * Ss * 2);
  float*    rlh = (float*)alloc((size_t)BHh * Ss * 32 * 4);
  float*    rlw = (float*)alloc((size_t)BHh * Ss * 32 * 4);
  _Float16* oh  = (_Float16*)alloc(M1 * Cc * 2);

  // 1) precision staging
  cvt_f16_kernel<<<(int)((M1 * Cc + 255) / 256), 256, 0, stream>>>(
      x, xh, (int)(M1 * Cc));
  transpose_cvt_kernel<<<(Cc * 3 * Cc + 255) / 256, 256, 0, stream>>>(
      wqkv, wqT, Cc, 3 * Cc);
  transpose_cvt_kernel<<<(Cc * Cc + 255) / 256, 256, 0, stream>>>(
      wproj, wpT, Cc, Cc);

  // 2) QKV projection (WMMA), scatter to q(scaled)/k/v^T
  wmma_gemm_kernel<<<dim3(64, 36), 256, 0, stream>>>(
      xh, wqT, bqkv, (int)M1, 3 * Cc, Cc, 1, nullptr, qf, kf, vTf);

  // 3) decomposed relative position dots
  rel_bias_kernel<<<(BHh * Ss * 32) / 256, 256, 0, stream>>>(
      qf, rph, rpw, rlh, rlw);

  // 4) fused flash attention (WMMA QK^T + softmax + WMMA PV)
  flash_attn_kernel<<<dim3(16, 96), 128, 0, stream>>>(
      qf, kf, vTf, rlh, rlw, oh);

  // 5) output projection (WMMA), f32 out + bias
  wmma_gemm_kernel<<<dim3(64, 12), 256, 0, stream>>>(
      oh, wpT, bproj, (int)M1, Cc, Cc, 0, out, nullptr, nullptr, nullptr);
}
